// CausalSTDiT2Block_7653631722023
// MI455X (gfx1250) — matrix operand order repaired and checked by reading, not compiled, not run on hardware
//
#include <hip/hip_runtime.h>
#include <hip/hip_bf16.h>

// ---------------- problem constants ----------------
#define C_   1152
#define NH_  16
#define HD_  72
#define B_   4
#define T_   16
#define S_   256
#define N_   4096           // T_*S_
#define LY_  120
#define SCALE_ 0.1178511301977579f   // HD^-0.5

typedef __attribute__((ext_vector_type(16))) _Float16 v16h;
typedef __attribute__((ext_vector_type(8)))  _Float16 v8h;
typedef __attribute__((ext_vector_type(8)))  float    v8f;

__device__ __forceinline__ v8f wmma_f16(v16h a, v16h b, v8f c) {
  return __builtin_amdgcn_wmma_f32_16x16x32_f16(false, a, false, b, (short)0, c, false, false);
}

// A fragment (16x32 f16) per ISA 7.12.2: lane l: row m = l&15, hi=l>>4;
// halves 0..7 -> K = hi*8 + j ; halves 8..15 -> K = 16 + hi*8 + (j-8)
__device__ __forceinline__ v16h lds_a_frag(const _Float16* base, int ld, int m0, int kc, int lane) {
  int hi = lane >> 4;
  const _Float16* p = base + (size_t)(m0 + (lane & 15)) * ld + kc;
  v8h lo = *(const v8h*)(p + hi * 8);
  v8h hh = *(const v8h*)(p + 16 + hi * 8);
  return __builtin_shufflevector(lo, hh, 0,1,2,3,4,5,6,7,8,9,10,11,12,13,14,15);
}
// B fragment (32x16 f16): lane l: col n = l&15, hi=l>>4; halves 0..7 -> K = hi*8 + j ; halves 8..15 -> K = 16 + hi*8 + (j-8).
// Requires B staged transposed in LDS as [n][k] (k contiguous).
__device__ __forceinline__ v16h lds_b_frag(const _Float16* base, int ld, int n0, int kc, int lane) {
  int hi = lane >> 4;
  const _Float16* p = base + (size_t)(n0 + (lane & 15)) * ld + kc + hi * 8;
  v8h lo = *(const v8h*)(p);
  v8h hh = *(const v8h*)(p + 16);
  return __builtin_shufflevector(lo, hh, 0,1,2,3,4,5,6,7,8,9,10,11,12,13,14,15);
}

__device__ __forceinline__ v8f v8f_zero() {
  v8f z;
  #pragma unroll
  for (int i = 0; i < 8; ++i) z[i] = 0.f;
  return z;
}

// ---------------- generic GEMM: C = act(A @ W + bias) ----------------
// A[M,K] fp32 row-major, W[K,Nn] fp32 row-major, Cout[M,Nn] fp32.
// N,K multiples of 128/32; M guarded. act: 0 none, 1 gelu(tanh).
#define TM 128
#define TN 128
#define TK 32
__global__ __launch_bounds__(256) void gemm_bias_act(
    const float* __restrict__ A, const float* __restrict__ W,
    const float* __restrict__ bias, float* __restrict__ Cout,
    int M, int Nn, int K, int act)
{
  __shared__ __align__(16) _Float16 As[TM][TK];
  __shared__ __align__(16) _Float16 Bs[TN][TK];   // transposed: [n][k]
  const int tid  = threadIdx.x;
  const int lane = tid & 31;
  const int wave = tid >> 5;
  const int wm = wave >> 2;        // 0..1  (rows)
  const int wn = wave & 3;         // 0..3  (cols)
  const int m0 = blockIdx.y * TM;
  const int n0 = blockIdx.x * TN;

  v8f acc[4][2];
  #pragma unroll
  for (int i = 0; i < 4; ++i)
    #pragma unroll
    for (int j = 0; j < 2; ++j) acc[i][j] = v8f_zero();

  for (int k0 = 0; k0 < K; k0 += TK) {
    #pragma unroll
    for (int it = 0; it < 16; ++it) {
      int i = tid + it * 256;               // 0..4095
      int r = i >> 5, c = i & 31;
      float v = (m0 + r < M) ? A[(size_t)(m0 + r) * K + k0 + c] : 0.f;
      As[r][c] = (_Float16)v;
    }
    #pragma unroll
    for (int it = 0; it < 16; ++it) {
      int i = tid + it * 256;
      int kk = i >> 7, nn = i & 127;
      Bs[nn][kk] = (_Float16)W[(size_t)(k0 + kk) * Nn + n0 + nn];
    }
    __syncthreads();
    v16h af[4], bf[2];
    #pragma unroll
    for (int mt = 0; mt < 4; ++mt) af[mt] = lds_a_frag(&As[0][0], TK, wm * 64 + mt * 16, 0, lane);
    #pragma unroll
    for (int nt = 0; nt < 2; ++nt) bf[nt] = lds_b_frag(&Bs[0][0], TK, wn * 32 + nt * 16, 0, lane);
    #pragma unroll
    for (int mt = 0; mt < 4; ++mt)
      #pragma unroll
      for (int nt = 0; nt < 2; ++nt)
        acc[mt][nt] = wmma_f16(af[mt], bf[nt], acc[mt][nt]);
    __syncthreads();
  }

  const int hi = lane >> 4, ln = lane & 15;
  #pragma unroll
  for (int mt = 0; mt < 4; ++mt)
    #pragma unroll
    for (int nt = 0; nt < 2; ++nt)
      #pragma unroll
      for (int r = 0; r < 8; ++r) {
        int gm = m0 + wm * 64 + mt * 16 + r + 8 * hi;
        int gn = n0 + wn * 32 + nt * 16 + ln;
        if (gm < M) {
          float v = acc[mt][nt][r] + bias[gn];
          if (act == 1) {
            float u = v;
            v = 0.5f * u * (1.f + tanhf(0.7978845608028654f * (u + 0.044715f * u * u * u)));
          }
          Cout[(size_t)gm * Nn + gn] = v;
        }
      }
}

// ---------------- generic attention (no mask / no rope) ----------------
// One wave per 16-query tile per (seq,head). Head dim padded 72->96,
// keys padded to NT*16 (masked). 128 threads = 4 waves per block.
template <int NT>
__global__ __launch_bounds__(128) void attn_generic(
    const float* __restrict__ Qp, int q_seq_rows, int q_ld, int q_coff,
    const float* __restrict__ KVp, int kv_seq_rows, int kv_ld, int k_coff, int v_coff,
    int Lkv, float* __restrict__ Op, int out_ld)
{
  __shared__ __align__(16) _Float16 Ps[4][16][NT * 16];
  const int lane = threadIdx.x & 31;
  const int wave = threadIdx.x >> 5;
  const int u = blockIdx.x;
  const int seq = u / NH_;
  const int h = u % NH_;
  const int qtile = blockIdx.y * 4 + wave;
  const int hi = lane >> 4, ln = lane & 15;

  const float* Qb = Qp + ((size_t)seq * q_seq_rows + (size_t)qtile * 16) * q_ld + q_coff + h * HD_;
  const float* Kb = KVp + (size_t)seq * kv_seq_rows * kv_ld + k_coff + h * HD_;
  const float* Vb = KVp + (size_t)seq * kv_seq_rows * kv_ld + v_coff + h * HD_;

  // Q A-fragments for 3 depth chunks (96 = 3*32, zero-padded past 72)
  v16h aq[3];
  #pragma unroll
  for (int kc = 0; kc < 3; ++kc) {
    #pragma unroll
    for (int j = 0; j < 8; ++j) {
      int d = kc * 32 + hi * 8 + j;
      aq[kc][j] = (d < HD_) ? (_Float16)Qb[(size_t)ln * q_ld + d] : (_Float16)0.f;
    }
    #pragma unroll
    for (int j = 0; j < 8; ++j) {
      int d = kc * 32 + 16 + hi * 8 + j;
      aq[kc][8 + j] = (d < HD_) ? (_Float16)Qb[(size_t)ln * q_ld + d] : (_Float16)0.f;
    }
  }

  // Scores S = Q K^T
  v8f sacc[NT];
  #pragma unroll
  for (int nt = 0; nt < NT; ++nt) sacc[nt] = v8f_zero();
  #pragma unroll
  for (int nt = 0; nt < NT; ++nt) {
    int key = nt * 16 + ln;
    #pragma unroll
    for (int kc = 0; kc < 3; ++kc) {
      v16h bk;
      #pragma unroll
      for (int j = 0; j < 16; ++j) {
        int d = kc * 32 + (j < 8 ? hi * 8 + j : 16 + hi * 8 + (j - 8));
        bk[j] = (key < Lkv && d < HD_) ? (_Float16)Kb[(size_t)key * kv_ld + d] : (_Float16)0.f;
      }
      sacc[nt] = wmma_f16(aq[kc], bk, sacc[nt]);
    }
  }

  // scale + key mask + row softmax (row m = r + 8*hi, cols across 16-lane half)
  float mx[8], sm[8];
  #pragma unroll
  for (int r = 0; r < 8; ++r) mx[r] = -3.0e38f;
  #pragma unroll
  for (int nt = 0; nt < NT; ++nt) {
    int key = nt * 16 + ln;
    #pragma unroll
    for (int r = 0; r < 8; ++r) {
      float v = sacc[nt][r] * SCALE_;
      if (key >= Lkv) v = -1.0e30f;
      sacc[nt][r] = v;
      mx[r] = fmaxf(mx[r], v);
    }
  }
  #pragma unroll
  for (int r = 0; r < 8; ++r) {
    #pragma unroll
    for (int off = 1; off < 16; off <<= 1) mx[r] = fmaxf(mx[r], __shfl_xor(mx[r], off));
  }
  #pragma unroll
  for (int r = 0; r < 8; ++r) sm[r] = 0.f;
  #pragma unroll
  for (int nt = 0; nt < NT; ++nt)
    #pragma unroll
    for (int r = 0; r < 8; ++r) {
      float e = __expf(sacc[nt][r] - mx[r]);
      sacc[nt][r] = e;
      sm[r] += e;
    }
  #pragma unroll
  for (int r = 0; r < 8; ++r) {
    #pragma unroll
    for (int off = 1; off < 16; off <<= 1) sm[r] += __shfl_xor(sm[r], off);
    sm[r] = 1.f / sm[r];
  }
  #pragma unroll
  for (int nt = 0; nt < NT; ++nt)
    #pragma unroll
    for (int r = 0; r < 8; ++r)
      Ps[wave][r + 8 * hi][nt * 16 + ln] = (_Float16)(sacc[nt][r] * sm[r]);

  // O = P @ V  (depth 80 -> 5 n-tiles, keys in NT/2 chunks of 32)
  v8f oacc[5];
  #pragma unroll
  for (int vt = 0; vt < 5; ++vt) oacc[vt] = v8f_zero();
  #pragma unroll
  for (int kc2 = 0; kc2 < NT / 2; ++kc2) {
    v16h pa = lds_a_frag(&Ps[wave][0][0], NT * 16, 0, kc2 * 32, lane);
    #pragma unroll
    for (int vt = 0; vt < 5; ++vt) {
      v16h bv;
      int hd = vt * 16 + ln;
      #pragma unroll
      for (int j = 0; j < 16; ++j) {
        int key = kc2 * 32 + (j < 8 ? hi * 8 + j : 16 + hi * 8 + (j - 8));
        bv[j] = (key < Lkv && hd < HD_) ? (_Float16)Vb[(size_t)key * kv_ld + hd] : (_Float16)0.f;
      }
      oacc[vt] = wmma_f16(pa, bv, oacc[vt]);
    }
  }
  #pragma unroll
  for (int vt = 0; vt < 5; ++vt)
    #pragma unroll
    for (int r = 0; r < 8; ++r) {
      int col = vt * 16 + ln;
      if (col < HD_) {
        size_t gm = (size_t)seq * q_seq_rows + (size_t)qtile * 16 + r + 8 * hi;
        Op[gm * out_ld + h * HD_ + col] = oacc[vt][r];
      }
    }
}

// ---------------- temporal attention: causal + RoPE, L=T_=16 ----------------
// One wave per (b,s,head). 256 threads = 8 waves; grid (B*S, 2).
__global__ __launch_bounds__(256) void attn_temporal(
    const float* __restrict__ qkv, const float* __restrict__ cosT,
    const float* __restrict__ sinT, float* __restrict__ Op)
{
  __shared__ __align__(16) _Float16 Ps[8][16][32];
  const int lane = threadIdx.x & 31;
  const int wave = threadIdx.x >> 5;
  const int seq = blockIdx.x;            // 0..B*S-1
  const int b = seq >> 8;                // / S_
  const int s = seq & (S_ - 1);
  const int h = blockIdx.y * 8 + wave;   // 0..15
  const int hi = lane >> 4, ln = lane & 15;
  const size_t ld3 = 3 * C_;
  const float* base = qkv + ((size_t)b * N_ + s) * ld3;   // + t*S_*ld3

  auto elem = [&](int tt, int d, int coff) -> float {
    if (d >= HD_) return 0.f;
    const float* row = base + (size_t)tt * S_ * ld3 + coff + h * HD_;
    float xe = row[d & ~1], xo = row[d | 1];
    float c = cosT[tt * (HD_ / 2) + (d >> 1)];
    float sn = sinT[tt * (HD_ / 2) + (d >> 1)];
    return (d & 1) ? (xe * sn + xo * c) : (xe * c - xo * sn);
  };

  v16h aq[3], bk[3];
  #pragma unroll
  for (int kc = 0; kc < 3; ++kc) {
    #pragma unroll
    for (int j = 0; j < 8; ++j)  aq[kc][j]     = (_Float16)elem(ln, kc * 32 + hi * 8 + j, 0);
    #pragma unroll
    for (int j = 0; j < 8; ++j)  aq[kc][8 + j] = (_Float16)elem(ln, kc * 32 + 16 + hi * 8 + j, 0);
    #pragma unroll
    for (int j = 0; j < 16; ++j) bk[kc][j]     = (_Float16)elem(ln, kc * 32 + (j < 8 ? hi * 8 + j : 16 + hi * 8 + (j - 8)), C_);
  }

  v8f sacc = v8f_zero();
  #pragma unroll
  for (int kc = 0; kc < 3; ++kc) sacc = wmma_f16(aq[kc], bk[kc], sacc);

  // scale + causal mask + softmax
  float p[8];
  float mx[8], sm[8];
  #pragma unroll
  for (int r = 0; r < 8; ++r) {
    int m = r + 8 * hi;
    float v = sacc[r] * SCALE_;
    if (ln > m) v = -1.0e30f;
    p[r] = v;
    mx[r] = v;
  }
  #pragma unroll
  for (int r = 0; r < 8; ++r) {
    #pragma unroll
    for (int off = 1; off < 16; off <<= 1) mx[r] = fmaxf(mx[r], __shfl_xor(mx[r], off));
  }
  #pragma unroll
  for (int r = 0; r < 8; ++r) { p[r] = __expf(p[r] - mx[r]); sm[r] = p[r]; }
  #pragma unroll
  for (int r = 0; r < 8; ++r) {
    #pragma unroll
    for (int off = 1; off < 16; off <<= 1) sm[r] += __shfl_xor(sm[r], off);
    p[r] /= sm[r];
  }

  // P -> per-wave LDS (16x32, cols 16..31 zero)
  for (int i = lane; i < 16 * 32; i += 32) (&Ps[wave][0][0])[i] = (_Float16)0.f;
  #pragma unroll
  for (int r = 0; r < 8; ++r) Ps[wave][r + 8 * hi][ln] = (_Float16)p[r];

  // O = P @ V
  v8f oacc[5];
  #pragma unroll
  for (int vt = 0; vt < 5; ++vt) oacc[vt] = v8f_zero();
  v16h pa = lds_a_frag(&Ps[wave][0][0], 32, 0, 0, lane);
  #pragma unroll
  for (int vt = 0; vt < 5; ++vt) {
    v16h bv;
    int hd = vt * 16 + ln;
    #pragma unroll
    for (int j = 0; j < 16; ++j) {
      int key = (j < 8 ? hi * 8 + j : 16 + hi * 8 + (j - 8));
      bv[j] = (key < T_ && hd < HD_)
                ? (_Float16)base[(size_t)key * S_ * ld3 + 2 * C_ + h * HD_ + hd]
                : (_Float16)0.f;
    }
    oacc[vt] = wmma_f16(pa, bv, oacc[vt]);
  }
  #pragma unroll
  for (int vt = 0; vt < 5; ++vt)
    #pragma unroll
    for (int r = 0; r < 8; ++r) {
      int col = vt * 16 + ln;
      int tq = r + 8 * hi;
      if (col < HD_)
        Op[((size_t)b * N_ + (size_t)tq * S_ + s) * C_ + h * HD_ + col] = oacc[vt][r];
    }
}

// ---------------- elementwise kernels ----------------
__global__ __launch_bounds__(256) void k_ss(const float* __restrict__ tab,
                                            const float* __restrict__ t,
                                            float* __restrict__ ss, int n) {
  int i = blockIdx.x * 256 + threadIdx.x;
  if (i < n) ss[i] = tab[i % (6 * C_)] + t[i];
}

__global__ __launch_bounds__(256) void k_copy(const float* __restrict__ src,
                                              float* __restrict__ dst, size_t n) {
  size_t i = (size_t)blockIdx.x * 256 + threadIdx.x;
  if (i < n) dst[i] = src[i];
}

__global__ __launch_bounds__(256) void k_ln_mod(const float* __restrict__ x,
                                                const float* __restrict__ ss,
                                                int shift_idx, int scale_idx,
                                                float* __restrict__ dst) {
  const int row = blockIdx.x;
  const int b = row / N_;
  const float* xr = x + (size_t)row * C_;
  float s1 = 0.f, s2 = 0.f;
  for (int i = threadIdx.x; i < C_; i += 256) { float v = xr[i]; s1 += v; s2 += v * v; }
  #pragma unroll
  for (int off = 16; off > 0; off >>= 1) { s1 += __shfl_down(s1, off); s2 += __shfl_down(s2, off); }
  __shared__ float r1[8], r2[8];
  __shared__ float mu_s, rs_s;
  const int lane = threadIdx.x & 31, w = threadIdx.x >> 5;
  if (lane == 0) { r1[w] = s1; r2[w] = s2; }
  __syncthreads();
  if (threadIdx.x == 0) {
    float a = 0.f, q = 0.f;
    for (int i = 0; i < 8; ++i) { a += r1[i]; q += r2[i]; }
    float mu = a / (float)C_;
    float var = q / (float)C_ - mu * mu;
    mu_s = mu;
    rs_s = rsqrtf(var + 1e-6f);
  }
  __syncthreads();
  const float mu = mu_s, rs = rs_s;
  const float* sh = ss + ((size_t)b * 6 + shift_idx) * C_;
  const float* sc = ss + ((size_t)b * 6 + scale_idx) * C_;
  for (int i = threadIdx.x; i < C_; i += 256)
    dst[(size_t)row * C_ + i] = (xr[i] - mu) * rs * (1.f + sc[i]) + sh[i];
}

__global__ __launch_bounds__(256) void k_resgate(float* __restrict__ x,
                                                 const float* __restrict__ src,
                                                 const float* __restrict__ ss,
                                                 int gidx, size_t total) {
  size_t i = (size_t)blockIdx.x * 256 + threadIdx.x;
  if (i >= total) return;
  int c = (int)(i % C_);
  int b = (int)(i / ((size_t)N_ * C_));
  float g = (gidx >= 0) ? ss[((size_t)b * 6 + gidx) * C_ + c] : 1.0f;
  x[i] += g * src[i];
}

// ---------------- driver ----------------
extern "C" void kernel_launch(void* const* d_in, const int* in_sizes, int n_in,
                              void* d_out, int out_size, void* d_ws, size_t ws_size,
                              hipStream_t stream) {
  const float* x        = (const float*)d_in[0];
  const float* y        = (const float*)d_in[1];
  const float* t        = (const float*)d_in[2];
  const float* sst      = (const float*)d_in[3];
  const float* w_qkv_s  = (const float*)d_in[4];
  const float* b_qkv_s  = (const float*)d_in[5];
  const float* w_proj_s = (const float*)d_in[6];
  const float* b_proj_s = (const float*)d_in[7];
  const float* w_qkv_t  = (const float*)d_in[8];
  const float* b_qkv_t  = (const float*)d_in[9];
  const float* w_proj_t = (const float*)d_in[10];
  const float* b_proj_t = (const float*)d_in[11];
  const float* wq_c     = (const float*)d_in[12];
  const float* bq_c     = (const float*)d_in[13];
  const float* wkv_c    = (const float*)d_in[14];
  const float* bkv_c    = (const float*)d_in[15];
  const float* wo_c     = (const float*)d_in[16];
  const float* bo_c     = (const float*)d_in[17];
  const float* w_fc1    = (const float*)d_in[18];
  const float* b_fc1    = (const float*)d_in[19];
  const float* w_fc2    = (const float*)d_in[20];
  const float* b_fc2    = (const float*)d_in[21];
  const float* fcos     = (const float*)d_in[22];
  const float* fsin     = (const float*)d_in[23];
  (void)in_sizes; (void)n_in; (void)out_size; (void)ws_size;

  float* xo = (float*)d_out;                 // running x (B,N,C) = final output
  float* ws = (float*)d_ws;
  float* ss      = ws;                                        // B*6*C
  float* bufXm   = ss      + (size_t)B_ * 6 * C_;             // B*N*C
  float* bufBig  = bufXm   + (size_t)B_ * N_ * C_;            // B*N*4C (qkv / mlp hidden)
  float* bufAttn = bufBig  + (size_t)B_ * N_ * 4 * C_;        // B*N*C
  float* bufP    = bufAttn + (size_t)B_ * N_ * C_;            // B*N*C
  float* bufKV   = bufP    + (size_t)B_ * N_ * C_;            // B*LY*2C

  const size_t BNC = (size_t)B_ * N_ * C_;
  const int M = B_ * N_;                     // 16384 rows
  const dim3 blk256(256), blk128(128);
  const int eg = (int)((BNC + 255) / 256);

  // ss = table + t
  k_ss<<<dim3((B_ * 6 * C_ + 255) / 256), blk256, 0, stream>>>(sst, t, ss, B_ * 6 * C_);
  // running x = input x
  k_copy<<<dim3(eg), blk256, 0, stream>>>(x, xo, BNC);
  // x_m = modulate(LN(x), shift_msa, scale_msa)
  k_ln_mod<<<dim3(M), blk256, 0, stream>>>(xo, ss, 0, 1, bufXm);
  // spatial qkv
  gemm_bias_act<<<dim3(27, 128), blk256, 0, stream>>>(bufXm, w_qkv_s, b_qkv_s, bufBig, M, 3 * C_, C_, 0);
  // spatial attention (64 seqs x 256, no mask)
  attn_generic<16><<<dim3(B_ * T_ * NH_, 4), blk128, 0, stream>>>(
      bufBig, S_, 3 * C_, 0, bufBig, S_, 3 * C_, C_, 2 * C_, S_, bufAttn, C_);
  // spatial proj + gated residual
  gemm_bias_act<<<dim3(9, 128), blk256, 0, stream>>>(bufAttn, w_proj_s, b_proj_s, bufP, M, C_, C_, 0);
  k_resgate<<<dim3(eg), blk256, 0, stream>>>(xo, bufP, ss, 2, BNC);
  // temporal qkv (on raw x)
  gemm_bias_act<<<dim3(27, 128), blk256, 0, stream>>>(xo, w_qkv_t, b_qkv_t, bufBig, M, 3 * C_, C_, 0);
  // temporal attention (causal + RoPE)
  attn_temporal<<<dim3(B_ * S_, 2), blk256, 0, stream>>>(bufBig, fcos, fsin, bufAttn);
  // temporal proj + gated residual
  gemm_bias_act<<<dim3(9, 128), blk256, 0, stream>>>(bufAttn, w_proj_t, b_proj_t, bufP, M, C_, C_, 0);
  k_resgate<<<dim3(eg), blk256, 0, stream>>>(xo, bufP, ss, 2, BNC);
  // cross attention: q from x, kv from y
  gemm_bias_act<<<dim3(9, 128), blk256, 0, stream>>>(xo, wq_c, bq_c, bufP, M, C_, C_, 0);
  gemm_bias_act<<<dim3(18, 4), blk256, 0, stream>>>(y, wkv_c, bkv_c, bufKV, B_ * LY_, 2 * C_, C_, 0);
  attn_generic<8><<<dim3(B_ * NH_, 64), blk128, 0, stream>>>(
      bufP, N_, C_, 0, bufKV, LY_, 2 * C_, 0, C_, LY_, bufAttn, C_);
  gemm_bias_act<<<dim3(9, 128), blk256, 0, stream>>>(bufAttn, wo_c, bo_c, bufXm, M, C_, C_, 0);
  k_resgate<<<dim3(eg), blk256, 0, stream>>>(xo, bufXm, ss, -1, BNC);
  // MLP
  k_ln_mod<<<dim3(M), blk256, 0, stream>>>(xo, ss, 3, 4, bufXm);
  gemm_bias_act<<<dim3(36, 128), blk256, 0, stream>>>(bufXm, w_fc1, b_fc1, bufBig, M, 4 * C_, C_, 1);
  gemm_bias_act<<<dim3(9, 128), blk256, 0, stream>>>(bufBig, w_fc2, b_fc2, bufP, M, C_, 4 * C_, 0);
  k_resgate<<<dim3(eg), blk256, 0, stream>>>(xo, bufP, ss, 5, BNC);
}
